// SelfConnectionIntro_36962488549809
// MI455X (gfx1250) — matrix operand order repaired and checked by reading, not compiled, not run on hardware
//
#include <hip/hip_runtime.h>
#include <hip/hip_bf16.h>

// ---- types matching the gfx1250 WMMA builtins ----
typedef _Float16 v16h __attribute__((ext_vector_type(16)));
typedef _Float16 v8h  __attribute__((ext_vector_type(8)));
typedef float    v8f  __attribute__((ext_vector_type(8)));
typedef float    v4f  __attribute__((ext_vector_type(4)));

#define Z_TOTAL 200000
#define DIMX    480
#define NSPEC   16

// ------------------------------------------------------------------
// CDNA5 async global->LDS copy (ASYNCcnt) with compile-safe fallback.
// Probe-learned signature: param0 is "int __vector_size__(16) addrspace(1)*".
// ------------------------------------------------------------------
#if defined(__HIP_DEVICE_COMPILE__) && defined(__gfx1250__) && \
    __has_builtin(__builtin_amdgcn_global_load_async_to_lds_b128)
#define FCTP_ASYNC_LDS 1
#endif

#ifdef FCTP_ASYNC_LDS
typedef int fctp_v4i __attribute__((vector_size(16)));
typedef __attribute__((address_space(1))) fctp_v4i* fctp_g4i;
typedef __attribute__((address_space(3))) fctp_v4i* fctp_l4i;
__device__ __forceinline__ void fctp_cp16_async(const void* g, void* l) {
    // 16 bytes, memory -> LDS, tracked on ASYNCcnt
    __builtin_amdgcn_global_load_async_to_lds_b128(
        (fctp_g4i)(uintptr_t)g,
        (fctp_l4i)(uint32_t)(uintptr_t)l,   // LDS offset = low 32 bits of generic addr
        0, 0);
}
__device__ __forceinline__ void fctp_wait_async() {
    asm volatile("s_wait_asynccnt 0x0" ::: "memory");
}
#endif

// Scheduler pinning: force DS-read lookahead ahead of each WMMA group.
// masks: 0x8 = MFMA/WMMA group, 0x100 = DS read
#if defined(__HIP_DEVICE_COMPILE__) && __has_builtin(__builtin_amdgcn_sched_group_barrier)
#define FCTP_SGB(mask, size) __builtin_amdgcn_sched_group_barrier((mask), (size), 0)
#else
#define FCTP_SGB(mask, size)
#endif

__device__ __forceinline__ v16h fctp_cat(v8h lo, v8h hi) {
    return __builtin_shufflevector(lo, hi, 0,1,2,3,4,5,6,7,8,9,10,11,12,13,14,15);
}
__device__ __forceinline__ v16h fctp_scale(v16h f, _Float16 s) {
#pragma unroll
    for (int i = 0; i < 16; ++i) f[i] *= s;   // v_pk_mul_f16
    return f;
}

// ------------------------------------------------------------------
// Prepack W[u][v][w] (f32) -> f16 B-fragments, contiguous per staged slice:
//   idx = ((((v*NT + nt)*UC + c)*2 + hf)*32 + K)*8 + n8
// lane K reads 2x16B (N=K&15; hf=0: k=8*(K>>4)+0..7, hf=1: k=16+8*(K>>4)+0..7) -> bank-friendly ds_b128.
// ------------------------------------------------------------------
__global__ void fctp_prepack_kernel(const float* __restrict__ W,
                                    _Float16* __restrict__ Bp,
                                    int U, int WN) {
    const int UC = U >> 5, NT = WN >> 4;
    const int total = NSPEC * NT * UC * 32 * 16;
    int idx = blockIdx.x * blockDim.x + threadIdx.x;
    if (idx >= total) return;
    int n8 = idx & 7;
    int K  = (idx >> 3) & 31;
    int hf = (idx >> 8) & 1;
    int t  = idx >> 9;
    int c  = t % UC; t /= UC;
    int nt = t % NT; t /= NT;
    int v  = t;
    int u  = c * 32 + hf * 16 + (K >> 4) * 8 + n8;
    int w  = nt * 16 + (K & 15);
    Bp[idx] = (_Float16)W[((size_t)u * NSPEC + v) * WN + w];
}

// ------------------------------------------------------------------
// One CG path: out[z,w,k] = alpha * sum_{v,u} op[z,v] x[z,u,k] W[u,v,w]
// 4 waves/WG (wave32), wave owns 16 z-rows x U cols x D components.
// B slices staged to LDS (double-buffered, async); B fragments software-
// pipelined (rotating 4-deep buffer + sched_group_barrier pinning).
// ------------------------------------------------------------------
template <int U, int D, int XOFF, int HALVES>
__global__ __launch_bounds__(128) void fctp_path_kernel(
    const float* __restrict__ x, const float* __restrict__ op,
    const _Float16* __restrict__ Bp, float* __restrict__ out, float alpha)
{
    constexpr int UC     = U / 32;          // K chunks of 32 per species
    constexpr int NT     = U / 16;          // output col tiles
    constexpr int NTH    = NT / HALVES;     // col tiles per staged slice
    constexpr int PITCH  = U + 8;           // f16 pitch: 16B align + no bank conflicts
    constexpr int SLICE  = NTH * UC * 512;  // f16 per staged B slice
    constexpr int NSTEPS = NSPEC * HALVES;
    constexpr int F      = NTH * UC;        // B fragments per step
    constexpr int NBUF   = (F < 4) ? F : 4; // rotating fragment buffer depth
    constexpr int PRE    = (F < 2) ? F : 2; // preload depth

    __shared__ alignas(16) _Float16 Bbuf[2][SLICE];
    __shared__ alignas(16) _Float16 xs[D][64][PITCH];
    __shared__ float opsz[64][NSPEC];

    const int zbase = blockIdx.x * 64;
    const int tid   = threadIdx.x;          // 0..127
    const int wave  = tid >> 5;             // 16-row group
    const int lane  = tid & 31;

    // ---- stage x tile (de-interleave to [kk][row][u], f32->f16) ----
    if constexpr (D == 1) {
        for (int e8 = tid; e8 < 64 * U / 8; e8 += 128) {
            int row = (e8 * 8) / U, u0 = (e8 * 8) % U;
            const float* src = x + (size_t)(zbase + row) * DIMX + XOFF + u0;
            v4f a = *(const v4f*)src;
            v4f b = *(const v4f*)(src + 4);
            v8h h;
#pragma unroll
            for (int i = 0; i < 4; ++i) { h[i] = (_Float16)a[i]; h[4 + i] = (_Float16)b[i]; }
            *(v8h*)&xs[0][row][u0] = h;
        }
    } else {
        for (int e4 = tid; e4 < 64 * U * D / 4; e4 += 128) {
            int row = (e4 * 4) / (U * D), r0 = (e4 * 4) % (U * D);
            v4f a = *(const v4f*)(x + (size_t)(zbase + row) * DIMX + XOFF + r0);
#pragma unroll
            for (int i = 0; i < 4; ++i) {
                int r = r0 + i;
                xs[r % D][row][r / D] = (_Float16)a[i];
            }
        }
    }
    // ---- stage operand * alpha ----
    for (int e = tid; e < 64 * NSPEC; e += 128)
        opsz[e >> 4][e & 15] = op[(size_t)(zbase + (e >> 4)) * NSPEC + (e & 15)] * alpha;

    // ---- kick off first B slice ----
#ifdef FCTP_ASYNC_LDS
    for (int i = tid; i < SLICE / 8; i += 128)
        fctp_cp16_async(Bp + (size_t)i * 8, &Bbuf[0][i * 8]);
#else
    for (int i = tid; i < SLICE / 8; i += 128)
        *(v8h*)&Bbuf[0][i * 8] = *(const v8h*)(Bp + (size_t)i * 8);
#endif
    __syncthreads();   // xs / opsz visible

    const int m  = lane & 15;
    const int rl = wave * 16 + m;
    const int s0 = (lane >> 4) * 8;

    // A fragment: lane<16 holds K0-7,16-23; lane>=16 holds K8-15,24-31
    auto loadA = [&](int kk, int c) -> v16h {
        const _Float16* base = &xs[kk][rl][c * 32 + s0];
        return fctp_cat(*(const v8h*)base, *(const v8h*)(base + 16));
    };
    // B fragment: N col = lane&15; halves hold K0-7,16-23 (lane<16) / K8-15,24-31 (lane>=16)
    auto loadB = [&](const _Float16* buf, int f) -> v16h {
        const _Float16* bb = buf + f * 512 + lane * 8;
        return fctp_cat(*(const v8h*)bb, *(const v8h*)(bb + 256));
    };

    v8f acc[D][NT];
#pragma unroll
    for (int kk = 0; kk < D; ++kk)
#pragma unroll
        for (int nt = 0; nt < NT; ++nt)
#pragma unroll
            for (int i = 0; i < 8; ++i) acc[kk][nt][i] = 0.0f;

    // ---- main pipeline: (A reload+scale) -> wait -> barrier -> stage(next) -> wmma ----
#pragma unroll 1
    for (int v = 0; v < NSPEC; ++v) {
        // xs/opsz are immutable after the prologue: issue these ds reads before
        // the barrier so their latency hides under the barrier wait.
        const _Float16 sc = (_Float16)opsz[rl][v];
        v16h A[D][UC];
#pragma unroll
        for (int kk = 0; kk < D; ++kk)
#pragma unroll
            for (int c = 0; c < UC; ++c) A[kk][c] = fctp_scale(loadA(kk, c), sc);

#pragma unroll
        for (int h = 0; h < HALVES; ++h) {
            const int s = v * HALVES + h;
#ifdef FCTP_ASYNC_LDS
            fctp_wait_async();            // own stage(s) complete
#endif
            __syncthreads();              // everyone's stage(s) complete
            if (s + 1 < NSTEPS) {         // overlap: stage next slice during compute
                const _Float16* src = Bp + (size_t)(s + 1) * SLICE;
                _Float16* dst = &Bbuf[(s + 1) & 1][0];
#ifdef FCTP_ASYNC_LDS
                for (int i = tid; i < SLICE / 8; i += 128)
                    fctp_cp16_async(src + (size_t)i * 8, dst + i * 8);
#else
                for (int i = tid; i < SLICE / 8; i += 128)
                    *(v8h*)(dst + i * 8) = *(const v8h*)(src + (size_t)i * 8);
#endif
            }
            const _Float16* buf = &Bbuf[s & 1][0];

            // software-pipelined B fragments: preload 2, stay 2 ahead
            v16h Bf[NBUF];
#pragma unroll
            for (int p = 0; p < PRE; ++p) Bf[p] = loadB(buf, p);
#pragma unroll
            for (int f = 0; f < F; ++f) {
                if (f + PRE < F) Bf[(f + PRE) % NBUF] = loadB(buf, f + PRE);
                const int ntl = f / UC, c = f % UC;
#pragma unroll
                for (int kk = 0; kk < D; ++kk) {
                    acc[kk][h * NTH + ntl] = __builtin_amdgcn_wmma_f32_16x16x32_f16(
                        false, A[kk][c], false, Bf[f % NBUF], (short)0,
                        acc[kk][h * NTH + ntl], false, false);
                }
            }

            // pin the schedule: 2 fragments (4 ds reads) of lookahead before
            // each WMMA group; consumes exactly 2F DS reads and F*D WMMAs.
            FCTP_SGB(0x100, 4);
#pragma unroll
            for (int f = 0; f < F - 2; ++f) {
                FCTP_SGB(0x8, D);
                FCTP_SGB(0x100, 2);
            }
            FCTP_SGB(0x8, 2 * D);
        }
    }

    // ---- epilogue: C layout -> VGPR r holds (M = r + 8*(lane>>4), N = lane&15) ----
    const int Mhi = (lane >> 4) * 8;
#pragma unroll
    for (int kk = 0; kk < D; ++kk)
#pragma unroll
        for (int nt = 0; nt < NT; ++nt)
#pragma unroll
            for (int r = 0; r < 8; ++r) {
                int M  = r + Mhi;
                int zz = zbase + wave * 16 + M;
                int w  = nt * 16 + (lane & 15);
                out[(size_t)zz * DIMX + XOFF + w * D + kk] = acc[kk][nt][r];
            }
}

// ------------------------------------------------------------------
// launch
// ------------------------------------------------------------------
extern "C" void kernel_launch(void* const* d_in, const int* in_sizes, int n_in,
                              void* d_out, int out_size, void* d_ws, size_t ws_size,
                              hipStream_t stream) {
    const float* x  = (const float*)d_in[0];   // [200000, 480]
    const float* op = (const float*)d_in[1];   // [200000, 16]
    const float* w0 = (const float*)d_in[2];   // [128, 16, 128]
    const float* w1 = (const float*)d_in[3];   // [64, 16, 64]
    const float* w2 = (const float*)d_in[4];   // [32, 16, 32]
    float* out = (float*)d_out;

    _Float16* B0 = (_Float16*)d_ws;                      // 512 KB
    _Float16* B1 = B0 + (size_t)NSPEC * 128 * 128;       // 128 KB
    _Float16* B2 = B1 + (size_t)NSPEC * 64 * 64;         //  32 KB

    const int n0 = NSPEC * 128 * 128;
    const int n1 = NSPEC * 64 * 64;
    const int n2 = NSPEC * 32 * 32;
    fctp_prepack_kernel<<<(n0 + 255) / 256, 256, 0, stream>>>(w0, B0, 128, 128);
    fctp_prepack_kernel<<<(n1 + 255) / 256, 256, 0, stream>>>(w1, B1, 64, 64);
    fctp_prepack_kernel<<<(n2 + 255) / 256, 256, 0, stream>>>(w2, B2, 32, 32);

    const int ZB = Z_TOTAL / 64;             // 3125, exact
    const float a0 = 0.022097086912079612f;  // 1/sqrt(128*16)
    const float a1 = 0.03125f;               // 1/sqrt(64*16)
    const float a2 = 0.044194173824159224f;  // 1/sqrt(32*16)
    fctp_path_kernel<128, 1,   0, 2><<<ZB, 128, 0, stream>>>(x, op, B0, out, a0);
    fctp_path_kernel< 64, 3, 128, 1><<<ZB, 128, 0, stream>>>(x, op, B1, out, a1);
    fctp_path_kernel< 32, 5, 320, 1><<<ZB, 128, 0, stream>>>(x, op, B2, out, a2);
}
